// KAAttention_22814866276846
// MI455X (gfx1250) — matrix operand order repaired and checked by reading, not compiled, not run on hardware
//
#include <hip/hip_runtime.h>
#include <hip/hip_bf16.h>
#include <math.h>

// ============================================================================
// KAAttention on MI455X (gfx1250, wave32, WMMA, async global->LDS DMA).
//
// Pipeline:
//   1) cvt fp32 -> bf16 (hidden, Wqkv, Wd)
//   2) WMMA GEMM: qkv = X @ Wqkv^T + bqkv     (double-buffered async staging)
//   3) RoPE + split into q_bf (scaled by HD^-0.5), k_bf, v_f32  [bh][s][d]
//   4) attention: per (b,h) serial over 16-row blocks:
//        scores (WMMA bf16) -> global scratch (L2-resident, 192MB L2)
//        row softmax (wave32 shfl reductions)
//        A_i = diag(P)_i * v_i + sum_{j<i} P_ij @ A_j   (history via WMMA,
//          transposed bf16 history layout for contiguous B-fragments)
//        16-step serial in-block triangular solve
//   5) WMMA GEMM: out = A @ Wd^T + bd
//
// Workspace requirement: ~122 MB.
// ============================================================================

constexpr int B_   = 2;
constexpr int S_   = 2048;
constexpr int HID_ = 1024;
constexpr int NH_  = 16;
constexpr int HD_  = 64;
constexpr int RD_  = 16;          // rotary dims
constexpr int BH_  = B_ * NH_;    // 32
constexpr int NBLK = S_ / 16;     // 128 row-blocks

typedef __attribute__((ext_vector_type(16))) __bf16 bf16x16;
typedef __attribute__((ext_vector_type(8)))  float  f32x8;

struct alignas(16) U128 { unsigned int x[4]; };

union FragBF {
  bf16x16 v;
  U128    u[2];
  __bf16  e[16];
};

__device__ __forceinline__ __bf16 f2bf(float f) {
  // round-to-nearest-even fp32 -> bf16
  unsigned u = __float_as_uint(f);
  unsigned r = (u + 0x7FFFu + ((u >> 16) & 1u)) >> 16;
  unsigned short h = (unsigned short)r;
  __bf16 out;
  __builtin_memcpy(&out, &h, 2);
  return out;
}

// CDNA5 async DMA: global memory -> LDS, 16B per active lane, no VGPR
// round-trip. Tracked by ASYNCcnt (CDNA5 ISA). LDS address = low
// 32 bits of the generic pointer (flat->LDS truncation, ISA 10.2).
__device__ __forceinline__ void async_b128(const void* g, void* l) {
  const unsigned lds = (unsigned)(unsigned long long)l;
  asm volatile("global_load_async_to_lds_b128 %0, %1, off"
               :: "v"(lds), "v"(g) : "memory");
}
__device__ __forceinline__ void wait_async0() {
  asm volatile("s_wait_asynccnt 0x0" ::: "memory");
}

// ---------------------------------------------------------------------------
__global__ __launch_bounds__(256) void cvt_f32_to_bf16(
    const float* __restrict__ src, unsigned short* __restrict__ dst_raw, int n) {
  __bf16* __restrict__ dst = reinterpret_cast<__bf16*>(dst_raw);
  int i = blockIdx.x * 256 + threadIdx.x;
  if (i < n) dst[i] = f2bf(src[i]);
}

// ---------------------------------------------------------------------------
// C[M][N] = A[M][K](bf16) @ B[N][K]^T(bf16) + bias[N], fp32 accumulate.
// Block tile 128x256, 8 waves (2 along M x 4 along N), 64x64 per wave
// (4x4 WMMA tiles, 128 acc VGPRs). K step 32. Double-buffered LDS filled
// by async global->LDS DMA so tile k+1 streams in behind the WMMAs of tile k.
// Assumes M%128==0, N%256==0, K%32==0 (true for all uses here).
__global__ __launch_bounds__(256) void gemm_bias_bf16(
    const unsigned short* __restrict__ A_raw,
    const unsigned short* __restrict__ B_raw,
    const float* __restrict__ bias,
    float* __restrict__ C, int M, int N, int K) {
  const __bf16* __restrict__ A  = reinterpret_cast<const __bf16*>(A_raw);
  const __bf16* __restrict__ Bm = reinterpret_cast<const __bf16*>(B_raw);

  __shared__ alignas(16) __bf16 As[2][128][32];   // 16 KB
  __shared__ alignas(16) __bf16 Bs[2][256][32];   // 32 KB

  const int tid   = threadIdx.x;
  const int lane  = tid & 31;
  const int w     = tid >> 5;
  const int waveM = w & 1;         // 2 waves along M
  const int waveN = w >> 1;        // 4 waves along N
  const int laneM = lane & 15;
  const int half  = lane >> 4;
  const int bm = blockIdx.y * 128;
  const int bn = blockIdx.x * 256;

  f32x8 acc[4][4] = {};

  // 24KB per K-step = 1536 x 16B chunks = 6 async b128 per thread.
  auto stage = [&](int buf, int k0) {
#pragma unroll
    for (int c = 0; c < 6; ++c) {
      const int chunk = tid + c * 256;
      if (chunk < 512) {                       // As: 128 rows x 4 x 16B
        const int row = chunk >> 2, q = chunk & 3;
        async_b128(&A[(size_t)(bm + row) * K + k0 + q * 8],
                   &As[buf][row][q * 8]);
      } else {                                 // Bs: 256 rows x 4 x 16B
        const int idx = chunk - 512;
        const int row = idx >> 2, q = idx & 3;
        async_b128(&Bm[(size_t)(bn + row) * K + k0 + q * 8],
                   &Bs[buf][row][q * 8]);
      }
    }
  };

  stage(0, 0);
  wait_async0();
  __syncthreads();

  int cur = 0;
  for (int k0 = 0; k0 < K; k0 += 32) {
    const bool more = (k0 + 32) < K;
    if (more) stage(cur ^ 1, k0 + 32);   // DMA next tile behind the math

    // A fragment (16x32): lane = row m; K dwords {0..3}+4h and {8..11}+4h.
    // B fragment (32x16): lane = col n; K dwords {0..3}+4h and {8..11}+4h.
    FragBF af[4], bfm[4];
#pragma unroll
    for (int t = 0; t < 4; ++t) {
      const U128* pa = (const U128*)&As[cur][waveM * 64 + t * 16 + laneM][0];
      af[t].u[0] = pa[half];
      af[t].u[1] = pa[2 + half];
      const U128* pb = (const U128*)&Bs[cur][waveN * 64 + t * 16 + laneM][0];
      bfm[t].u[0] = pb[half];
      bfm[t].u[1] = pb[2 + half];
    }
#pragma unroll
    for (int tm = 0; tm < 4; ++tm)
#pragma unroll
      for (int tn = 0; tn < 4; ++tn)
        acc[tm][tn] = __builtin_amdgcn_wmma_f32_16x16x32_bf16(
            false, af[tm].v, false, bfm[tn].v, (short)0, acc[tm][tn],
            false, false);

    if (more) wait_async0();
    __syncthreads();
    cur ^= 1;
  }

  // C/D layout: element r -> row (r + 8*half), col lane%16
#pragma unroll
  for (int tm = 0; tm < 4; ++tm)
#pragma unroll
    for (int tn = 0; tn < 4; ++tn) {
      const int gn = bn + waveN * 64 + tn * 16 + laneM;
      const float bv = bias[gn];
#pragma unroll
      for (int r = 0; r < 8; ++r) {
        const int gm = bm + waveM * 64 + tm * 16 + r + half * 8;
        C[(size_t)gm * N + gn] = acc[tm][tn][r] + bv;
      }
    }
}

// ---------------------------------------------------------------------------
// RoPE on first RD dims + split qkv into per-head layouts.
__global__ __launch_bounds__(256) void rope_split_kernel(
    const float* __restrict__ qkv,            // [B*S][3*HID], [3][NH][HD] cols
    unsigned short* __restrict__ qbf_raw,     // [BH][S][HD], scaled by HD^-1/2
    unsigned short* __restrict__ kbf_raw,     // [BH][S][HD]
    float* __restrict__ vf) {                 // [BH][S][HD]
  __bf16* __restrict__ qbf = reinterpret_cast<__bf16*>(qbf_raw);
  __bf16* __restrict__ kbf = reinterpret_cast<__bf16*>(kbf_raw);
  int idx = blockIdx.x * 256 + threadIdx.x;   // over BH*S*HD
  if (idx >= BH_ * S_ * HD_) return;
  const int d  = idx & 63;
  const int s  = (idx >> 6) & (S_ - 1);
  const int bh = idx >> 17;
  const int h = bh & 15, b = bh >> 4;

  const size_t row = ((size_t)(b * S_ + s)) * (3 * HID_);
  const int co = h * HD_ + d;
  float q = qkv[row + co];
  float k = qkv[row + HID_ + co];
  const float v = qkv[row + 2 * HID_ + co];

  if (d < RD_) {
    const int j = d & 7;
    const float inv = powf(10000.0f, -(float)(2 * j) / (float)RD_);
    const float ang = (float)s * inv;
    const float cs = cosf(ang), sn = sinf(ang);
    if (d < 8) {
      q = q * cs - qkv[row + co + 8] * sn;
      k = k * cs - qkv[row + HID_ + co + 8] * sn;
    } else {
      q = q * cs + qkv[row + co - 8] * sn;
      k = k * cs + qkv[row + HID_ + co - 8] * sn;
    }
  }
  const size_t o = ((size_t)bh * S_ + s) * HD_ + d;
  qbf[o] = f2bf(q * 0.125f);   // fold HD^-0.5 logit scale into q
  kbf[o] = f2bf(k);
  vf[o]  = v;
}

// ---------------------------------------------------------------------------
// One workgroup (8 waves) per (b,h). Serial over 16-row blocks.
__global__ __launch_bounds__(256) void kaattention_kernel(
    const unsigned short* __restrict__ qbf_raw,
    const unsigned short* __restrict__ kbf_raw,
    const float* __restrict__ vf,
    float* __restrict__ probsws,              // [BH][16][S] scratch (L2)
    unsigned short* __restrict__ Abf_raw,     // [B][S][HID] for out-proj
    unsigned short* __restrict__ ATbf_raw) {  // [BH][HD][S] history (transposed)
  const __bf16* __restrict__ qbf = reinterpret_cast<const __bf16*>(qbf_raw);
  const __bf16* __restrict__ kbf = reinterpret_cast<const __bf16*>(kbf_raw);
  __bf16* __restrict__ Abf  = reinterpret_cast<__bf16*>(Abf_raw);
  __bf16* __restrict__ ATbf = reinterpret_cast<__bf16*>(ATbf_raw);

  __shared__ alignas(16) __bf16 qs[16][64];
  __shared__ float part[8][16][64];   // per-wave recurrence partials (32KB)
  __shared__ float Ablk[16][64];
  __shared__ float pblk[16][16];      // in-block (diagonal) prob tile
  __shared__ float pdiag[16];

  const int bh = blockIdx.x;
  const int b = bh >> 4, h = bh & 15;
  const int tid = threadIdx.x;
  const int lane = tid & 31, w = tid >> 5;
  const int laneM = lane & 15, half = lane >> 4;

  float* __restrict__ prow = probsws + (size_t)bh * 16 * S_;
  const __bf16* __restrict__ krow  = kbf + (size_t)bh * S_ * HD_;
  const float*  __restrict__ vrow  = vf  + (size_t)bh * S_ * HD_;
  __bf16* __restrict__ ATrow = ATbf + (size_t)bh * HD_ * S_;

  for (int i = 0; i < NBLK; ++i) {
    // ---- stage q block (16x64 bf16 = 2KB) in LDS via async DMA
    {
      const __bf16* src = qbf + ((size_t)bh * S_ + i * 16) * HD_;
      if (tid < 128) async_b128(src + tid * 8, &((__bf16*)qs)[tid * 8]);
      wait_async0();
    }
    __syncthreads();

    // ---- phase 1: logits tiles; wave w handles col-blocks j = w, w+8, ...
    FragBF aq[2];
#pragma unroll
    for (int c = 0; c < 2; ++c) {
      const U128* p = (const U128*)&qs[laneM][c * 32];
      aq[c].u[0] = p[half];
      aq[c].u[1] = p[2 + half];
    }
    for (int j = w; j <= i; j += 8) {
      f32x8 acc = {};
#pragma unroll
      for (int c = 0; c < 2; ++c) {
        FragBF bk;
        const U128* p = (const U128*)(krow +
            ((size_t)(j * 16 + laneM)) * HD_ + c * 32);
        bk.u[0] = p[half];
        bk.u[1] = p[2 + half];
        acc = __builtin_amdgcn_wmma_f32_16x16x32_bf16(
            false, aq[c].v, false, bk.v, (short)0, acc, false, false);
      }
#pragma unroll
      for (int r = 0; r < 8; ++r) {
        const int qr = r + half * 8;                 // query row in block
        float x = acc[r];
        if (j == i && laneM > qr) x = -1e30f;        // causal mask (diag block)
        prow[(size_t)qr * S_ + j * 16 + laneM] = x;
      }
    }
    __threadfence();      // make logits visible across waves (stores land in L2)
    __syncthreads();

    // ---- phase 2: softmax, 2 rows per wave, wave32 shfl reductions
    const int ncol = (i + 1) * 16;
    for (int rr = 2 * w; rr < 2 * w + 2; ++rr) {
      float* rp = prow + (size_t)rr * S_;
      float mx = -1e30f;
      for (int c = lane; c < ncol; c += 32) mx = fmaxf(mx, rp[c]);
      for (int o = 16; o >= 1; o >>= 1) mx = fmaxf(mx, __shfl_xor(mx, o, 32));
      float sum = 0.f;
      for (int c = lane; c < ncol; c += 32) {
        const float e = __expf(rp[c] - mx);
        rp[c] = e;
        sum += e;
      }
      for (int o = 16; o >= 1; o >>= 1) sum += __shfl_xor(sum, o, 32);
      const float inv = 1.0f / sum;
      for (int c = lane; c < ncol; c += 32) rp[c] *= inv;
      if (lane < 16) {
        const float pv = rp[i * 16 + lane];   // diagonal block columns
        pblk[rr][lane] = pv;
        if (lane == rr) pdiag[rr] = pv;
      }
    }
    __threadfence();
    __syncthreads();

    // ---- phase 3: history part  sum_{j<i} P_ij(16x16) @ A_j(16x64), bf16 WMMA
    // pairs of 16-col blocks give K=32; zero-pad the half past j==i-1.
    f32x8 pacc[4] = {};
    const int npair = (i + 1) >> 1;
    for (int p = w; p < npair; p += 8) {
      const int j0 = 2 * p;
      FragBF pa;   // A operand: P rows (convert fp32 probs -> bf16)
      const float* pr = prow + (size_t)laneM * S_ + j0 * 16;
#pragma unroll
      for (int v = 0; v < 8; ++v) {
        const int kk = 2 * v + half * 16;
        const bool ok = (j0 * 16 + kk) < i * 16;     // strict j < i
        pa.e[2 * v]     = f2bf(ok ? pr[kk]     : 0.f);
        pa.e[2 * v + 1] = f2bf(ok ? pr[kk + 1] : 0.f);
      }
      const bool hiOK = (j0 + 1) < i;   // B operand K=16..31 <-> half==1
#pragma unroll
      for (int nt = 0; nt < 4; ++nt) {
        FragBF hb;   // B operand: A_hist^T rows (hd-major, contiguous in seq)
        if (half == 0 || hiOK) {
          const U128* pb = (const U128*)(ATrow +
              ((size_t)(nt * 16 + laneM)) * S_ + j0 * 16 + half * 16);
          hb.u[0] = pb[0];
          hb.u[1] = pb[1];
        } else {
          U128 z{};
          hb.u[0] = z;
          hb.u[1] = z;
        }
        pacc[nt] = __builtin_amdgcn_wmma_f32_16x16x32_bf16(
            false, pa.v, false, hb.v, (short)0, pacc[nt], false, false);
      }
    }
#pragma unroll
    for (int nt = 0; nt < 4; ++nt)
#pragma unroll
      for (int r = 0; r < 8; ++r)
        part[w][r + half * 8][nt * 16 + laneM] = pacc[nt][r];
    __syncthreads();

    // ---- phase 4: deterministic reduction of wave partials + rhs = diag*v
    for (int idx = tid; idx < 16 * 64; idx += 256) {
      const int r = idx >> 6, d = idx & 63;
      float sv = pdiag[r] * vrow[(size_t)(i * 16 + r) * HD_ + d];
#pragma unroll
      for (int ww = 0; ww < 8; ++ww) sv += part[ww][r][d];
      Ablk[r][d] = sv;
    }
    __syncthreads();

    // ---- phase 5: serial in-block strict-lower triangular recurrence
    for (int r = 1; r < 16; ++r) {
      if (tid < 64) {
        float sv = 0.f;
        for (int c = 0; c < r; ++c) sv += pblk[r][c] * Ablk[c][tid];
        Ablk[r][tid] += sv;
      }
      __syncthreads();
    }

    // ---- phase 6: commit block -> [b,s,hid] (out-proj) and [bh,d,s] (history)
    for (int idx = tid; idx < 16 * 64; idx += 256) {
      const int r = idx >> 6, d = idx & 63;
      const __bf16 bv = f2bf(Ablk[r][d]);
      Abf[((size_t)(b * S_ + i * 16 + r)) * HID_ + h * HD_ + d] = bv;
      ATrow[(size_t)d * S_ + i * 16 + r] = bv;
    }
    __threadfence();   // history must be L2-visible before next block's phase 3
    __syncthreads();
  }
}

// ---------------------------------------------------------------------------
extern "C" void kernel_launch(void* const* d_in, const int* in_sizes, int n_in,
                              void* d_out, int out_size, void* d_ws,
                              size_t ws_size, hipStream_t stream) {
  (void)in_sizes; (void)n_in; (void)out_size; (void)ws_size;

  const float* hs   = (const float*)d_in[0];
  const float* Wqkv = (const float*)d_in[1];
  const float* bqkv = (const float*)d_in[2];
  const float* Wd   = (const float*)d_in[3];
  const float* bd   = (const float*)d_in[4];

  char* ws = (char*)d_ws;
  size_t off = 0;
  auto alloc = [&](size_t bytes) -> void* {
    void* p = ws + off;
    off += (bytes + 255) & ~(size_t)255;
    return p;
  };

  unsigned short* Xbf    = (unsigned short*)alloc((size_t)B_ * S_ * HID_ * 2);
  unsigned short* Wqkvbf = (unsigned short*)alloc((size_t)3 * HID_ * HID_ * 2);
  unsigned short* Wdbf   = (unsigned short*)alloc((size_t)HID_ * HID_ * 2);
  float*          qkvf   = (float*)alloc((size_t)B_ * S_ * 3 * HID_ * 4);
  unsigned short* qbf    = (unsigned short*)alloc((size_t)BH_ * S_ * HD_ * 2);
  unsigned short* kbf    = (unsigned short*)alloc((size_t)BH_ * S_ * HD_ * 2);
  float*          vfp    = (float*)alloc((size_t)BH_ * S_ * HD_ * 4);
  float*          probs  = (float*)alloc((size_t)BH_ * 16 * S_ * 4);
  unsigned short* Abf    = (unsigned short*)alloc((size_t)B_ * S_ * HID_ * 2);
  unsigned short* ATbf   = (unsigned short*)alloc((size_t)BH_ * HD_ * S_ * 2);

  const int nX = B_ * S_ * HID_;
  cvt_f32_to_bf16<<<(nX + 255) / 256, 256, 0, stream>>>(hs, Xbf, nX);
  const int nWq = 3 * HID_ * HID_;
  cvt_f32_to_bf16<<<(nWq + 255) / 256, 256, 0, stream>>>(Wqkv, Wqkvbf, nWq);
  const int nWd = HID_ * HID_;
  cvt_f32_to_bf16<<<(nWd + 255) / 256, 256, 0, stream>>>(Wd, Wdbf, nWd);

  // qkv = X @ Wqkv^T + bqkv : M=4096, N=3072, K=1024
  gemm_bias_bf16<<<dim3(3 * HID_ / 256, B_ * S_ / 128), 256, 0, stream>>>(
      Xbf, Wqkvbf, bqkv, qkvf, B_ * S_, 3 * HID_, HID_);

  const int nR = BH_ * S_ * HD_;
  rope_split_kernel<<<(nR + 255) / 256, 256, 0, stream>>>(qkvf, qbf, kbf, vfp);

  kaattention_kernel<<<BH_, 256, 0, stream>>>(qbf, kbf, vfp, probs, Abf, ATbf);

  // out = A @ Wd^T + bd : M=4096, N=1024, K=1024
  gemm_bias_bf16<<<dim3(HID_ / 256, B_ * S_ / 128), 256, 0, stream>>>(
      Abf, Wdbf, bd, (float*)d_out, B_ * S_, HID_, HID_);
}
